// Decoder_11424613007807
// MI455X (gfx1250) — matrix operand order repaired and checked by reading, not compiled, not run on hardware
//
#include <hip/hip_runtime.h>
#include <math.h>

// ---------------------------------------------------------------------------
// Model constants (match reference)
// ---------------------------------------------------------------------------
#define DMODEL 1024
#define NHEAD  16
#define DHEAD  64
#define SEQ    1024
#define BATCH  2
#define NLAYER 3
#define FFDIM  4096
#define VOCAB  32000

typedef __attribute__((ext_vector_type(16))) __bf16 v16bf;
typedef __attribute__((ext_vector_type(8)))  float  v8f;
typedef int v4i __attribute__((vector_size(16)));     // <4 x i32> for async b128 ops

#define AS_GLOBAL __attribute__((address_space(1)))
#define AS_LDS    __attribute__((address_space(3)))

// Async global->LDS copies (CDNA5 GLOBAL_LOAD_ASYNC_TO_LDS_B128, ASYNCcnt),
// guarded so the source compiles on toolchains without the builtin.
// Probe-confirmed signature: (v4i addrspace(1)*, v4i addrspace(3)*, imm offset, cpol)
#if defined(__has_builtin)
#  if __has_builtin(__builtin_amdgcn_global_load_async_to_lds_b128)
#    define HAVE_ASYNC_LDS 1
#  endif
#endif
#ifndef HAVE_ASYNC_LDS
#  define HAVE_ASYNC_LDS 0
#endif

#if HAVE_ASYNC_LDS
#  if __has_builtin(__builtin_amdgcn_s_wait_asynccnt)
#    define ASYNC_WAIT() __builtin_amdgcn_s_wait_asynccnt(0)
#  else
#    define ASYNC_WAIT() asm volatile("s_wait_asynccnt 0" ::: "memory")
#  endif
#  define ASYNC_CP_B128(gsrc, ldst, off) \
     __builtin_amdgcn_global_load_async_to_lds_b128( \
         (v4i AS_GLOBAL*)(gsrc), (v4i AS_LDS*)(ldst), (off), 0)
#else
#  define ASYNC_WAIT()
#endif

union Frag {
    v16bf v;
    unsigned int u[8];
};

// ---------------------------------------------------------------------------
// WMMA fragment loaders (CDNA5 ISA 16-bit layouts, wave32)
//   A 16x32  : lane m=l&15, g=l>>4, vgpr v holds K = (v/4)*16 + g*8 + (v%4)*2 (+1)
//   B 32x16  : stored transposed in LDS Bt[n][k]; lane n=l&15, vgpr v holds
//              K = (v/4)*16 + g*8 + (v%4)*2 (+1)  -> one contiguous u32 per VGPR
//   C/D 16x16: vgpr r holds row m = r + 8*g, col n = l&15
// ---------------------------------------------------------------------------
__device__ __forceinline__ void load_a_frag(const __bf16* base, int stride, int lane, Frag& f) {
    const int m = lane & 15, g = lane >> 4;
#pragma unroll
    for (int r = 0; r < 8; ++r) {
        const int kk = ((r >> 2) << 4) + (g << 3) + ((r & 3) << 1);
        f.u[r] = *(const unsigned int*)(base + m * stride + kk);
    }
}

__device__ __forceinline__ void load_bt_frag(const __bf16* base, int stride, int lane, Frag& f) {
    const int n = lane & 15, g = lane >> 4;
#pragma unroll
    for (int r = 0; r < 8; ++r) {
        const int kk = ((r >> 2) << 4) + (g << 3) + ((r & 3) << 1);
        f.u[r] = *(const unsigned int*)(base + n * stride + kk);
    }
}

__device__ __forceinline__ v8f wmma_bf16(const Frag& a, const Frag& b, v8f c) {
    return __builtin_amdgcn_wmma_f32_16x16x32_bf16(false, a.v, false, b.v, (short)0, c, false, false);
}

__device__ __forceinline__ float rmax16(float x) {
#pragma unroll
    for (int m = 1; m < 16; m <<= 1) x = fmaxf(x, __shfl_xor(x, m, 32));
    return x;
}
__device__ __forceinline__ float rsum16(float x) {
#pragma unroll
    for (int m = 1; m < 16; m <<= 1) x += __shfl_xor(x, m, 32);
    return x;
}

// ---------------------------------------------------------------------------
// GEMM: C[M,N] = A[M,K](bf16) x B[K,N](fp32, converted to bf16 in LDS)
// Block tile 128x64, 8 waves of 32x32 each, K-step 32, double-buffered LDS.
// A tile staged via async global->LDS copies when available; next B tile is
// fetched into registers before the current tile's WMMAs, stored after.
// Epilogue: optional bias[n], residual[m,n], ReLU; outputs fp32 and/or bf16.
// M%128==0, N%64==0, K%32==0 for all call sites here.
// ---------------------------------------------------------------------------
__global__ __launch_bounds__(256)
void gemm_bf16f32(const __bf16* __restrict__ A, const float* __restrict__ Bw,
                  const float* __restrict__ bias, const float* __restrict__ resid,
                  float* __restrict__ Cf, __bf16* __restrict__ Cb,
                  int M, int N, int K, int relu)
{
    __shared__ __align__(16) __bf16 As [2][128][32];
    __shared__ __align__(16) __bf16 BsT[2][64][32];

    const int tid  = threadIdx.x;
    const int lane = tid & 31;
    const int wave = tid >> 5;
    const int wm   = wave >> 1;       // 0..3
    const int wn   = wave & 1;        // 0..1
    const int m0   = blockIdx.y * 128;
    const int n0   = blockIdx.x * 64;

    const v8f zero = {0.f, 0.f, 0.f, 0.f, 0.f, 0.f, 0.f, 0.f};
    v8f acc[2][2] = {{zero, zero}, {zero, zero}};

    // staging maps
    const int arow = tid >> 1, acol = (tid & 1) << 4;     // A: 128 rows x 32 cols bf16
    const int brow = tid >> 3, bcol = (tid & 7) << 3;     // B: 32 rows x 64 cols fp32

    const __bf16* aSrc = A + (size_t)(m0 + arow) * K + acol;    // + kt
    const float*  bSrc = Bw + (size_t)brow * N + n0 + bcol;     // + kt*N

    // ---- prologue: stage K-tile 0 into buffer 0
    {
#if HAVE_ASYNC_LDS
        ASYNC_CP_B128(aSrc, &As[0][arow][acol], 0);
        ASYNC_CP_B128(aSrc, &As[0][arow][acol], 16);
#else
        const uint4* s = (const uint4*)aSrc;
        uint4 w0 = s[0], w1 = s[1];
        *(uint4*)&As[0][arow][acol]     = w0;
        *(uint4*)&As[0][arow][acol + 8] = w1;
#endif
        float4 f0 = *(const float4*)bSrc;
        float4 f1 = *(const float4*)(bSrc + 4);
        BsT[0][bcol + 0][brow] = (__bf16)f0.x;
        BsT[0][bcol + 1][brow] = (__bf16)f0.y;
        BsT[0][bcol + 2][brow] = (__bf16)f0.z;
        BsT[0][bcol + 3][brow] = (__bf16)f0.w;
        BsT[0][bcol + 4][brow] = (__bf16)f1.x;
        BsT[0][bcol + 5][brow] = (__bf16)f1.y;
        BsT[0][bcol + 6][brow] = (__bf16)f1.z;
        BsT[0][bcol + 7][brow] = (__bf16)f1.w;
        ASYNC_WAIT();
    }
    __syncthreads();

    const int nkt = K >> 5;
    for (int it = 0; it < nkt; ++it) {
        const int  cur  = it & 1;
        const int  nxt  = cur ^ 1;
        const int  ktn  = (it + 1) << 5;
        const bool more = (it + 1) < nkt;

        // ---- issue next tile's global traffic before this tile's WMMAs
        uint4  a0n = {0, 0, 0, 0}, a1n = {0, 0, 0, 0};
        float4 f0n = {0, 0, 0, 0}, f1n = {0, 0, 0, 0};
        if (more) {
#if HAVE_ASYNC_LDS
            ASYNC_CP_B128(aSrc + ktn, &As[nxt][arow][acol], 0);
            ASYNC_CP_B128(aSrc + ktn, &As[nxt][arow][acol], 16);
#else
            const uint4* s = (const uint4*)(aSrc + ktn);
            a0n = s[0];
            a1n = s[1];
#endif
            const float* sp = bSrc + (size_t)ktn * N;
            f0n = *(const float4*)sp;
            f1n = *(const float4*)(sp + 4);
            if (it + 2 < nkt) {   // global_prefetch_b8 of the tile after next
                __builtin_prefetch(aSrc + ktn + 32, 0, 1);
                __builtin_prefetch(bSrc + (size_t)(ktn + 32) * N, 0, 1);
            }
        }

        // ---- compute current tile
        Frag a0, a1, b0, b1;
        load_a_frag (&As [cur][wm * 32     ][0], 32, lane, a0);
        load_a_frag (&As [cur][wm * 32 + 16][0], 32, lane, a1);
        load_bt_frag(&BsT[cur][wn * 32     ][0], 32, lane, b0);
        load_bt_frag(&BsT[cur][wn * 32 + 16][0], 32, lane, b1);

        acc[0][0] = wmma_bf16(a0, b0, acc[0][0]);
        acc[0][1] = wmma_bf16(a0, b1, acc[0][1]);
        acc[1][0] = wmma_bf16(a1, b0, acc[1][0]);
        acc[1][1] = wmma_bf16(a1, b1, acc[1][1]);

        // ---- write next tile into the other LDS buffer, then fence
        if (more) {
#if !HAVE_ASYNC_LDS
            *(uint4*)&As[nxt][arow][acol]     = a0n;
            *(uint4*)&As[nxt][arow][acol + 8] = a1n;
#endif
            BsT[nxt][bcol + 0][brow] = (__bf16)f0n.x;
            BsT[nxt][bcol + 1][brow] = (__bf16)f0n.y;
            BsT[nxt][bcol + 2][brow] = (__bf16)f0n.z;
            BsT[nxt][bcol + 3][brow] = (__bf16)f0n.w;
            BsT[nxt][bcol + 4][brow] = (__bf16)f1n.x;
            BsT[nxt][bcol + 5][brow] = (__bf16)f1n.y;
            BsT[nxt][bcol + 6][brow] = (__bf16)f1n.z;
            BsT[nxt][bcol + 7][brow] = (__bf16)f1n.w;
            ASYNC_WAIT();
            __syncthreads();
        }
    }

    // ---- epilogue
    const int g  = lane >> 4;
    const int cn = lane & 15;
#pragma unroll
    for (int ti = 0; ti < 2; ++ti) {
#pragma unroll
        for (int tj = 0; tj < 2; ++tj) {
#pragma unroll
            for (int r = 0; r < 8; ++r) {
                const int row = m0 + wm * 32 + ti * 16 + r + 8 * g;
                const int col = n0 + wn * 32 + tj * 16 + cn;
                float v = acc[ti][tj][r];
                if (bias)  v += bias[col];
                if (resid) v += resid[(size_t)row * N + col];
                if (relu)  v = fmaxf(v, 0.f);
                if (Cf) Cf[(size_t)row * N + col] = v;
                if (Cb) Cb[(size_t)row * N + col] = (__bf16)v;
            }
        }
    }
}

// ---------------------------------------------------------------------------
// Flash attention (bf16 in, bf16 out, f32 accumulate, online softmax).
// Grid: (Tq/128, H, B); 8 waves/block, each wave owns a 16-row q tile.
// K-block of 32 keys per iteration: Q*K^T via 4 WMMA, P*V via 4 WMMA.
// K tile staged with async global->LDS when available (straight byte copy);
// V must transpose through registers. Q/K/V layout: [B, T, H*DH].
// ---------------------------------------------------------------------------
__global__ __launch_bounds__(256)
void flash_attn(const __bf16* __restrict__ Q, const __bf16* __restrict__ Kv,
                const __bf16* __restrict__ Vv, __bf16* __restrict__ O,
                int Tq, int Tk, int causal)
{
    __shared__ __align__(16) __bf16 Ks [32][64];      // K tile, row-major == Bt layout for scores
    __shared__ __align__(16) __bf16 VsT[64][32];      // V tile transposed: VsT[d][key]
    __shared__ __align__(16) __bf16 Ps [8][16][32];   // per-wave P staging (D-layout -> A-layout)

    const int tid  = threadIdx.x;
    const int lane = tid & 31;
    const int wave = tid >> 5;
    const int g    = lane >> 4;
    const int cn   = lane & 15;
    const int h    = blockIdx.y;
    const int b    = blockIdx.z;
    const int q0   = blockIdx.x * 128 + wave * 16;

    // Q fragments for this wave's 16 rows (d 0..31 and 32..63), loaded once
    Frag aq0, aq1;
    const __bf16* qbase = Q + (size_t)(b * Tq + q0) * DMODEL + h * DHEAD;
    load_a_frag(qbase,      DMODEL, lane, aq0);
    load_a_frag(qbase + 32, DMODEL, lane, aq1);

    const v8f zero = {0.f, 0.f, 0.f, 0.f, 0.f, 0.f, 0.f, 0.f};
    v8f o0 = zero, o1 = zero, o2 = zero, o3 = zero;
    float rowmax[8], rowsum[8];
#pragma unroll
    for (int r = 0; r < 8; ++r) { rowmax[r] = -3.0e38f; rowsum[r] = 0.f; }

    const int krow = tid >> 3, kcol = (tid & 7) << 3;   // 32 x 64 staging map
    const int nkb  = Tk / 32;

    for (int kb = 0; kb < nkb; ++kb) {
        __syncthreads();
        {   // stage K tile (row-major) and V tile (transposed)
            const __bf16* sk = Kv + (size_t)(b * Tk + kb * 32 + krow) * DMODEL + h * DHEAD + kcol;
            const __bf16* sv = Vv + (size_t)(b * Tk + kb * 32 + krow) * DMODEL + h * DHEAD + kcol;
#if HAVE_ASYNC_LDS
            ASYNC_CP_B128(sk, &Ks[krow][kcol], 0);
            uint4 vq = *(const uint4*)sv;
#else
            uint4 kq = *(const uint4*)sk;
            uint4 vq = *(const uint4*)sv;
            *(uint4*)&Ks[krow][kcol] = kq;
#endif
            __bf16 tmp[8];
            *(uint4*)tmp = vq;
#pragma unroll
            for (int j = 0; j < 8; ++j) VsT[kcol + j][krow] = tmp[j];
            ASYNC_WAIT();
        }
        __syncthreads();

        // fully-masked block for this wave's q tile: skip compute, barriers already done
        if (causal && kb * 32 > q0 + 15) continue;

        // S = Q * K^T  (two 16x16 key tiles, K-dim 64 = 2 chained WMMAs each)
        v8f s0 = zero, s1 = zero;
        Frag bk;
        load_bt_frag(&Ks[0][0],   64, lane, bk); s0 = wmma_bf16(aq0, bk, s0);
        load_bt_frag(&Ks[0][32],  64, lane, bk); s0 = wmma_bf16(aq1, bk, s0);
        load_bt_frag(&Ks[16][0],  64, lane, bk); s1 = wmma_bf16(aq0, bk, s1);
        load_bt_frag(&Ks[16][32], 64, lane, bk); s1 = wmma_bf16(aq1, bk, s1);

        // online softmax update (row = r + 8*g matches both S and O fragment layouts)
#pragma unroll
        for (int r = 0; r < 8; ++r) {
            const int qg = q0 + r + 8 * g;
            float e0 = s0[r] * 0.125f;                 // 1/sqrt(DH)
            float e1 = s1[r] * 0.125f;
            if (causal) {
                if (kb * 32 + cn      > qg) e0 = -3.0e38f;
                if (kb * 32 + 16 + cn > qg) e1 = -3.0e38f;
            }
            const float mb   = rmax16(fmaxf(e0, e1));
            const float mnew = fmaxf(rowmax[r], mb);
            const float corr = __expf(rowmax[r] - mnew);
            rowmax[r] = mnew;
            const float p0 = __expf(e0 - mnew);
            const float p1 = __expf(e1 - mnew);
            rowsum[r] = rowsum[r] * corr + rsum16(p0 + p1);
            o0[r] *= corr; o1[r] *= corr; o2[r] *= corr; o3[r] *= corr;
            const int mrow = r + 8 * g;
            Ps[wave][mrow][cn]      = (__bf16)p0;
            Ps[wave][mrow][16 + cn] = (__bf16)p1;
        }

        // wave-private LDS staging: wait for our DS stores before cross-lane reads
        asm volatile("s_wait_dscnt 0" ::: "memory");

        Frag ap, bv;
        load_a_frag(&Ps[wave][0][0], 32, lane, ap);
        load_bt_frag(&VsT[ 0][0], 32, lane, bv); o0 = wmma_bf16(ap, bv, o0);
        load_bt_frag(&VsT[16][0], 32, lane, bv); o1 = wmma_bf16(ap, bv, o1);
        load_bt_frag(&VsT[32][0], 32, lane, bv); o2 = wmma_bf16(ap, bv, o2);
        load_bt_frag(&VsT[48][0], 32, lane, bv); o3 = wmma_bf16(ap, bv, o3);
    }

    // normalize and store
    __bf16* obase = O + (size_t)(b * Tq + q0) * DMODEL + h * DHEAD;
#pragma unroll
    for (int r = 0; r < 8; ++r) {
        const float inv = 1.f / rowsum[r];
        const size_t ro = (size_t)(r + 8 * g) * DMODEL;
        obase[ro +  0 + cn] = (__bf16)(o0[r] * inv);
        obase[ro + 16 + cn] = (__bf16)(o1[r] * inv);
        obase[ro + 32 + cn] = (__bf16)(o2[r] * inv);
        obase[ro + 48 + cn] = (__bf16)(o3[r] * inv);
    }
}

// ---------------------------------------------------------------------------
// LayerNorm: one block per row of D=1024, fp32 in -> bf16 out
// ---------------------------------------------------------------------------
__global__ __launch_bounds__(256)
void layernorm_bf16(const float* __restrict__ x, const float* __restrict__ gamma,
                    const float* __restrict__ beta, __bf16* __restrict__ out)
{
    __shared__ float s1[256];
    __shared__ float s2[256];
    const int row = blockIdx.x, tid = threadIdx.x;
    const float* xr = x + (size_t)row * DMODEL;
    const float a0 = xr[tid], a1 = xr[tid + 256], a2 = xr[tid + 512], a3 = xr[tid + 768];
    s1[tid] = a0 + a1 + a2 + a3;
    s2[tid] = a0 * a0 + a1 * a1 + a2 * a2 + a3 * a3;
    __syncthreads();
    for (int o = 128; o > 0; o >>= 1) {
        if (tid < o) { s1[tid] += s1[tid + o]; s2[tid] += s2[tid + o]; }
        __syncthreads();
    }
    const float mu   = s1[0] * (1.f / DMODEL);
    const float var  = s2[0] * (1.f / DMODEL) - mu * mu;
    const float rstd = rsqrtf(var + 1e-5f);
    __bf16* orow = out + (size_t)row * DMODEL;
    orow[tid]       = (__bf16)((a0 - mu) * rstd * gamma[tid]       + beta[tid]);
    orow[tid + 256] = (__bf16)((a1 - mu) * rstd * gamma[tid + 256] + beta[tid + 256]);
    orow[tid + 512] = (__bf16)((a2 - mu) * rstd * gamma[tid + 512] + beta[tid + 512]);
    orow[tid + 768] = (__bf16)((a3 - mu) * rstd * gamma[tid + 768] + beta[tid + 768]);
}

// ---------------------------------------------------------------------------
// Embedding gather + positional add
// ---------------------------------------------------------------------------
__global__ void embed_k(const int* __restrict__ idx, const float* __restrict__ tok,
                        const float* __restrict__ pos, float* __restrict__ x, int n)
{
    const int i = blockIdx.x * 256 + threadIdx.x;
    if (i >= n) return;
    const int d  = i % DMODEL;
    const int bt = i / DMODEL;
    const int t  = bt % SEQ;
    x[i] = tok[(size_t)idx[bt] * DMODEL + d] + pos[(size_t)t * DMODEL + d];
}

__global__ void f32_to_bf16(const float* __restrict__ in, __bf16* __restrict__ out, int n)
{
    const int i = blockIdx.x * 256 + threadIdx.x;
    if (i < n) out[i] = (__bf16)in[i];
}

// ---------------------------------------------------------------------------
// Cross-entropy: per-row logsumexp, then deterministic single-block reduce
// ---------------------------------------------------------------------------
__global__ __launch_bounds__(256)
void row_loss(const float* __restrict__ logits, const int* __restrict__ tgt,
              float* __restrict__ rl)
{
    __shared__ float sm[256];
    const int row = blockIdx.x, tid = threadIdx.x;
    const float* lr = logits + (size_t)row * VOCAB;
    float mx = -3.0e38f;
    for (int j = tid; j < VOCAB; j += 256) mx = fmaxf(mx, lr[j]);
    sm[tid] = mx; __syncthreads();
    for (int o = 128; o > 0; o >>= 1) {
        if (tid < o) sm[tid] = fmaxf(sm[tid], sm[tid + o]);
        __syncthreads();
    }
    mx = sm[0]; __syncthreads();
    float s = 0.f;
    for (int j = tid; j < VOCAB; j += 256) s += __expf(lr[j] - mx);
    sm[tid] = s; __syncthreads();
    for (int o = 128; o > 0; o >>= 1) {
        if (tid < o) sm[tid] += sm[tid + o];
        __syncthreads();
    }
    if (tid == 0) rl[row] = -(lr[tgt[row]] - mx - logf(sm[0]));
}

__global__ __launch_bounds__(256)
void loss_reduce(const float* __restrict__ rl, float* __restrict__ loss, int n)
{
    __shared__ float sm[256];
    const int tid = threadIdx.x;
    float s = 0.f;
    for (int j = tid; j < n; j += 256) s += rl[j];
    sm[tid] = s; __syncthreads();
    for (int o = 128; o > 0; o >>= 1) {
        if (tid < o) sm[tid] += sm[tid + o];
        __syncthreads();
    }
    if (tid == 0) loss[0] = sm[0] / (float)n;
}

// ---------------------------------------------------------------------------
// Orchestration
// d_in: 0 idx, 1 k, 2 v, 3 targets, 4 tok_emb, 5 pos_emb,
//       6..9 Wq_s Wk_s Wv_s Wo_s, 10..13 Wq_c Wk_c Wv_c Wo_c,
//       14 W1, 15 b1, 16 W2, 17 b2,
//       18/19 ln1_g/b, 20/21 ln2_g/b, 22/23 ln3_g/b, 24/25 lnf_g/b,
//       26 lm_W, 27 lm_b
// d_out: logits [B*T*V] fp32, then loss [1]
// ---------------------------------------------------------------------------
extern "C" void kernel_launch(void* const* d_in, const int* in_sizes, int n_in,
                              void* d_out, int out_size, void* d_ws, size_t ws_size,
                              hipStream_t stream)
{
    (void)in_sizes; (void)n_in; (void)out_size; (void)ws_size;

    const int*   idx     = (const int*)d_in[0];
    const float* kin     = (const float*)d_in[1];
    const float* vin     = (const float*)d_in[2];
    const int*   targets = (const int*)d_in[3];
    const float* tok     = (const float*)d_in[4];
    const float* pos     = (const float*)d_in[5];
    const float* Wq_s = (const float*)d_in[6],  *Wk_s = (const float*)d_in[7];
    const float* Wv_s = (const float*)d_in[8],  *Wo_s = (const float*)d_in[9];
    const float* Wq_c = (const float*)d_in[10], *Wk_c = (const float*)d_in[11];
    const float* Wv_c = (const float*)d_in[12], *Wo_c = (const float*)d_in[13];
    const float* W1 = (const float*)d_in[14], *b1 = (const float*)d_in[15];
    const float* W2 = (const float*)d_in[16], *b2 = (const float*)d_in[17];
    const float* ln1_g = (const float*)d_in[18], *ln1_b = (const float*)d_in[19];
    const float* ln2_g = (const float*)d_in[20], *ln2_b = (const float*)d_in[21];
    const float* ln3_g = (const float*)d_in[22], *ln3_b = (const float*)d_in[23];
    const float* lnf_g = (const float*)d_in[24], *lnf_b = (const float*)d_in[25];
    const float* lm_W = (const float*)d_in[26], *lm_b = (const float*)d_in[27];

    float* logits = (float*)d_out;
    float* loss   = logits + (size_t)BATCH * SEQ * VOCAB;

    const int M = BATCH * SEQ;               // 2048
    char* wp = (char*)d_ws;
    auto carve = [&](size_t bytes) -> void* {
        void* r = (void*)wp;
        wp += (bytes + 255) & ~(size_t)255;
        return r;
    };
    float*  x      = (float*) carve((size_t)M * DMODEL * 4);
    __bf16* hbuf   = (__bf16*)carve((size_t)M * DMODEL * 2);
    __bf16* qb     = (__bf16*)carve((size_t)M * DMODEL * 2);
    __bf16* kb     = (__bf16*)carve((size_t)M * DMODEL * 2);
    __bf16* vb     = (__bf16*)carve((size_t)M * DMODEL * 2);
    __bf16* attout = (__bf16*)carve((size_t)M * DMODEL * 2);
    __bf16* kinb   = (__bf16*)carve((size_t)M * DMODEL * 2);
    __bf16* vinb   = (__bf16*)carve((size_t)M * DMODEL * 2);
    __bf16* ff1    = (__bf16*)carve((size_t)M * FFDIM  * 2);
    float*  rl     = (float*) carve((size_t)M * 4);

    auto gemm = [&](const __bf16* A, const float* Bm, const float* bias,
                    const float* resid, float* Cf, __bf16* Cb,
                    int Mm, int Nn, int Kk, int relu) {
        dim3 grid(Nn / 64, Mm / 128);
        gemm_bf16f32<<<grid, 256, 0, stream>>>(A, Bm, bias, resid, Cf, Cb, Mm, Nn, Kk, relu);
    };

    const int nED = M * DMODEL;
    embed_k<<<(nED + 255) / 256, 256, 0, stream>>>(idx, tok, pos, x, nED);
    f32_to_bf16<<<(nED + 255) / 256, 256, 0, stream>>>(kin, kinb, nED);
    f32_to_bf16<<<(nED + 255) / 256, 256, 0, stream>>>(vin, vinb, nED);

    const dim3 agrid(SEQ / 128, NHEAD, BATCH);
    for (int i = 0; i < NLAYER; ++i) {
        const size_t wo = (size_t)i * DMODEL * DMODEL;
        // --- self attention ---
        layernorm_bf16<<<M, 256, 0, stream>>>(x, ln1_g + i * DMODEL, ln1_b + i * DMODEL, hbuf);
        gemm(hbuf, Wq_s + wo, nullptr, nullptr, nullptr, qb, M, DMODEL, DMODEL, 0);
        gemm(hbuf, Wk_s + wo, nullptr, nullptr, nullptr, kb, M, DMODEL, DMODEL, 0);
        gemm(hbuf, Wv_s + wo, nullptr, nullptr, nullptr, vb, M, DMODEL, DMODEL, 0);
        flash_attn<<<agrid, 256, 0, stream>>>(qb, kb, vb, attout, SEQ, SEQ, 1);
        gemm(attout, Wo_s + wo, nullptr, x, x, nullptr, M, DMODEL, DMODEL, 0);
        // --- cross attention ---
        layernorm_bf16<<<M, 256, 0, stream>>>(x, ln2_g + i * DMODEL, ln2_b + i * DMODEL, hbuf);
        gemm(hbuf, Wq_c + wo, nullptr, nullptr, nullptr, qb, M, DMODEL, DMODEL, 0);
        gemm(kinb, Wk_c + wo, nullptr, nullptr, nullptr, kb, M, DMODEL, DMODEL, 0);
        gemm(vinb, Wv_c + wo, nullptr, nullptr, nullptr, vb, M, DMODEL, DMODEL, 0);
        flash_attn<<<agrid, 256, 0, stream>>>(qb, kb, vb, attout, SEQ, SEQ, 0);
        gemm(attout, Wo_c + wo, nullptr, x, x, nullptr, M, DMODEL, DMODEL, 0);
        // --- MLP ---
        layernorm_bf16<<<M, 256, 0, stream>>>(x, ln3_g + i * DMODEL, ln3_b + i * DMODEL, hbuf);
        gemm(hbuf, W1 + (size_t)i * DMODEL * FFDIM, b1 + (size_t)i * FFDIM,
             nullptr, nullptr, ff1, M, FFDIM, DMODEL, 1);
        gemm(ff1, W2 + (size_t)i * FFDIM * DMODEL, b2 + (size_t)i * DMODEL,
             x, x, nullptr, M, DMODEL, FFDIM, 0);
    }

    layernorm_bf16<<<M, 256, 0, stream>>>(x, lnf_g, lnf_b, hbuf);
    gemm(hbuf, lm_W, lm_b, nullptr, logits, nullptr, M, VOCAB, DMODEL, 0);

    row_loss<<<M, 256, 0, stream>>>(logits, targets, rl);
    loss_reduce<<<1, 256, 0, stream>>>(rl, loss, M);
}
